// RelativePositionTransformer_90254442758646
// MI455X (gfx1250) — matrix operand order repaired and checked by reading, not compiled, not run on hardware
//
#include <hip/hip_runtime.h>
#include <math.h>

// ---------------------------------------------------------------------------
// VITS RelativePositionTransformer for MI455X (gfx1250, wave32, WMMA).
// All GEMM-shaped work runs on v_wmma_f32_16x16x32_f16 (fp32->fp16 converted
// while staging tiles into LDS, pre-swizzled into the ISA fragment layouts).
// All leading dimensions are compile-time template constants so staging loads
// become independent immediate-offset vector loads (no serialized address
// chains). LDS tiles are double-buffered (2 x 8 KB) to overlap the next
// tile's global loads with the current tile's WMMAs.
// Workspace requirement: ~119 MB (29,655,040 floats).
// ---------------------------------------------------------------------------

typedef _Float16 v16h __attribute__((ext_vector_type(16)));
typedef float    v8f  __attribute__((ext_vector_type(8)));

constexpr int LAYERS = 6;
constexpr int Bb = 4;      // batch
constexpr int Cc = 1024;   // channels
constexpr int Hh = 16;     // heads
constexpr int Dd = 64;     // head dim
constexpr int Tt = 512;    // sequence (== LDB == LDC everywhere)
constexpr int Ff = 4096;   // ffn channels
constexpr int KSc = 3;     // conv kernel
constexpr int WIN = 4;     // rel window

// ---------------------------------------------------------------------------
// fp32-in / fp16-WMMA / fp32-out GEMM, N = 512 fixed, LDB = LDC = 512 fixed.
//   OUT[b](M x 512) (+)= alpha * A[b](M x K) * B[b](K x 512)  (+ bias[m])
// A element (m,k):  !TA -> A[m*LDA + k*KDA]   (KDA=3 slices conv weights)
//                    TA -> A[k*LDA + m]
// B element (k,n):  !TB -> B[k*512 + n+shiftN] (zero outside [0,512))
//                    TB -> B[n*512 + k]
// Tile 64x64, BK=32 (one WMMA K-step), 4 waves (128 threads).
// M multiple of 64 (or exactly 64), K multiple of 32 -- true for every call.
// LDS tiles are stored pre-swizzled in WMMA fragment order:
//   A frag: lane L holds row m=(L&15), K pattern {kb..kb+7, 16+kb..16+kb+7},
//           kb = (L<16)?0:8   (CDNA5 ISA "16-bit A-Matrix 16x32")
//   B frag: lane L holds col n=(L&15), same K pattern as A frag
// so each lane fragment is 16 contiguous halves (32B) in LDS.
// ---------------------------------------------------------------------------
template <int LDA, int KDA, bool TA, bool TB, bool ACC>
__global__ void __launch_bounds__(128)
gemm_wmma_f16(const float* __restrict__ A, long sAb,
              const float* __restrict__ Bm, long sBb, int shiftN,
              float* __restrict__ Cm, long sCb,
              const float* __restrict__ bias, float alpha,
              int M, int K)
{
    __shared__ alignas(32) _Float16 As[2][64 * 32];
    __shared__ alignas(32) _Float16 Bs[2][64 * 32];

    const int tid  = threadIdx.x;
    const int lane = tid & 31;
    const int wave = tid >> 5;
    const int wm   = wave & 1;     // wave row (0..1)  -> 32 rows
    const int wn   = wave >> 1;    // wave col (0..1)  -> 32 cols

    const int m0 = blockIdx.y * 64;
    const int n0 = blockIdx.x * 64;
    const long zb = blockIdx.z;

    const float* __restrict__ Ab = A + zb * sAb;
    const float* __restrict__ Bg = Bm + zb * sBb;
    float* __restrict__ Cb = Cm + zb * sCb;

    // staging role: thread <-> (16-row/col sub-block, lane-in-fragment)
    const int blk  = tid >> 5;             // 0..3
    const int l15  = lane & 15;
    const int mloc = blk * 16 + l15;       // A row in tile
    const int nloc = blk * 16 + l15;       // B col in tile
    const int kb8  = (lane < 16) ? 0 : 8;  // A fragment half-K base
    const int kb16 = (lane < 16) ? 0 : 16; // unused; B staging uses kb8 like A

    // B column (fixed per thread) for the non-transposed path, clamped so the
    // load is always in-bounds; OOB lanes write 0 (branch-free, EXEC stays 1s).
    const int nsB   = n0 + nloc + shiftN;
    const bool okB  = (nsB >= 0) && (nsB < Tt);
    const int nsafe = okB ? nsB : (n0 + nloc);

    v8f acc[2][2];
#pragma unroll
    for (int f = 0; f < 2; ++f)
#pragma unroll
        for (int g = 0; g < 2; ++g)
#pragma unroll
            for (int r = 0; r < 8; ++r) acc[f][g][r] = 0.f;

    // ---- tile staging (all offsets compile-time constants off one base) ----
    auto stageA = [&](int k0, int buf) {
        _Float16* dst = &As[buf][(blk * 32 + lane) * 16];
        if (TA) {
            const float* Ap = Ab + (long)(k0 + kb8) * LDA + (m0 + mloc);
#pragma unroll
            for (int i = 0; i < 16; ++i) {
                const int kpat = ((i >> 3) << 4) + (i & 7);   // 0..7, 16..23
                dst[i] = (_Float16)Ap[(long)kpat * LDA];
            }
        } else {
            const float* Ap = Ab + (long)(m0 + mloc) * LDA + (long)(k0 + kb8) * KDA;
#pragma unroll
            for (int i = 0; i < 16; ++i) {
                const int kpat = ((i >> 3) << 4) + (i & 7);
                dst[i] = (_Float16)Ap[kpat * KDA];
            }
        }
    };
    auto stageB = [&](int k0, int buf) {
        _Float16* dst = &Bs[buf][(blk * 32 + lane) * 16];
        if (TB) {
            const float* Bp = Bg + (long)(n0 + nloc) * Tt + (k0 + kb8);
#pragma unroll
            for (int i = 0; i < 16; ++i) dst[i] = (_Float16)Bp[((i >> 3) << 4) + (i & 7)];
        } else {
            const float* Bp = Bg + (long)(k0 + kb8) * Tt + nsafe;
#pragma unroll
            for (int i = 0; i < 16; ++i) {
                const float v = Bp[(long)(((i >> 3) << 4) + (i & 7)) * Tt];
                dst[i] = (_Float16)(okB ? v : 0.f);
            }
        }
    };

    stageA(0, 0);
    stageB(0, 0);
    __syncthreads();

    int buf = 0;
    for (int k0 = 0; k0 < K; k0 += 32) {
        if (k0 + 32 < K) {                 // prefetch next tile into other buf
            stageA(k0 + 32, buf ^ 1);
            stageB(k0 + 32, buf ^ 1);
        }
        const v16h a0 = *reinterpret_cast<const v16h*>(&As[buf][((wm * 2 + 0) * 32 + lane) * 16]);
        const v16h a1 = *reinterpret_cast<const v16h*>(&As[buf][((wm * 2 + 1) * 32 + lane) * 16]);
        const v16h b0 = *reinterpret_cast<const v16h*>(&Bs[buf][((wn * 2 + 0) * 32 + lane) * 16]);
        const v16h b1 = *reinterpret_cast<const v16h*>(&Bs[buf][((wn * 2 + 1) * 32 + lane) * 16]);

        acc[0][0] = __builtin_amdgcn_wmma_f32_16x16x32_f16(false, a0, false, b0, (short)0, acc[0][0], false, false);
        acc[0][1] = __builtin_amdgcn_wmma_f32_16x16x32_f16(false, a0, false, b1, (short)0, acc[0][1], false, false);
        acc[1][0] = __builtin_amdgcn_wmma_f32_16x16x32_f16(false, a1, false, b0, (short)0, acc[1][0], false, false);
        acc[1][1] = __builtin_amdgcn_wmma_f32_16x16x32_f16(false, a1, false, b1, (short)0, acc[1][1], false, false);

        __syncthreads();
        buf ^= 1;
    }

    // ---- epilogue: 16x16 f32 C layout (VGPR r: m=r+8*(lane>=16), n=lane&15)
    const int nfr  = lane & 15;
    const int rofs = (lane >> 4) << 3;
#pragma unroll
    for (int f = 0; f < 2; ++f)
#pragma unroll
        for (int g = 0; g < 2; ++g) {
            const int mbase = m0 + wm * 32 + f * 16;
            const int nbase = n0 + wn * 32 + g * 16;
#pragma unroll
            for (int r = 0; r < 8; ++r) {
                const int m = mbase + rofs + r;
                const int n = nbase + nfr;
                float v = alpha * acc[f][g][r];
                const long idx = (long)m * Tt + n;
                if (ACC) {
                    Cb[idx] += v;
                } else {
                    if (bias) v += bias[m];
                    Cb[idx] = v;
                }
            }
        }
}

// ---------------------------------------------------------------------------
// R[b,h,t,r] = scale * sum_d q[b, h*64+d, t] * emb_k[r, d]   (r = 0..8)
// ---------------------------------------------------------------------------
__global__ void rel_logits_kernel(const float* __restrict__ q,
                                  const float* __restrict__ emb,
                                  float* __restrict__ R, float scale)
{
    const int idx = blockIdx.x * blockDim.x + threadIdx.x;   // B*H*T*9
    const int r   = idx % 9;
    const int bht = idx / 9;
    const int t   = bht % Tt;
    const int bh  = bht / Tt;
    const float* qp = q + (long)bh * Dd * Tt + t;
    const float* ep = emb + r * Dd;
    float acc = 0.f;
#pragma unroll 8
    for (int d = 0; d < Dd; ++d) acc += qp[(long)d * Tt] * ep[d];
    R[idx] = acc * scale;
}

// ---------------------------------------------------------------------------
// Row softmax with banded rel-logit add + mask.  One wave per row (T=512).
// ---------------------------------------------------------------------------
__global__ void __launch_bounds__(256)
softmax_rel_kernel(float* __restrict__ S, const float* __restrict__ R,
                   const float* __restrict__ mask)
{
    const int row  = blockIdx.x * 8 + (threadIdx.x >> 5);    // B*H*T rows
    const int lane = threadIdx.x & 31;
    const int bh   = row / Tt;
    const int t    = row - bh * Tt;
    const int b    = bh / Hh;
    float* Srow = S + (long)row * Tt;
    const float* Rrow = R + (long)row * 9;
    const float  mt   = mask[b * Tt + t];

    float x[16];
    float mx = -3.4e38f;
#pragma unroll
    for (int i = 0; i < 16; ++i) {
        const int s = lane + 32 * i;
        float v = Srow[s];
        const int r = s - t + WIN;
        if (r >= 0 && r < 2 * WIN + 1) v += Rrow[r];
        const float ms = mask[b * Tt + s];
        if (mt * ms == 0.f) v = -1e4f;
        x[i] = v;
        mx = fmaxf(mx, v);
    }
#pragma unroll
    for (int o = 16; o > 0; o >>= 1) mx = fmaxf(mx, __shfl_xor(mx, o, 32));
    float sum = 0.f;
#pragma unroll
    for (int i = 0; i < 16; ++i) { x[i] = __expf(x[i] - mx); sum += x[i]; }
#pragma unroll
    for (int o = 16; o > 0; o >>= 1) sum += __shfl_xor(sum, o, 32);
    const float inv = 1.f / sum;
#pragma unroll
    for (int i = 0; i < 16; ++i) Srow[lane + 32 * i] = x[i] * inv;
}

// ---------------------------------------------------------------------------
// attn[b, h*64+d, t] += sum_r P[b,h,t, t+r-4] * emb_v[r, d]
// ---------------------------------------------------------------------------
__global__ void add_relv_kernel(float* __restrict__ attn,
                                const float* __restrict__ P,
                                const float* __restrict__ embv)
{
    const long idx = (long)blockIdx.x * blockDim.x + threadIdx.x;  // B*C*T
    const int  t   = (int)(idx % Tt);
    const long ct  = idx / Tt;
    const int  c   = (int)(ct % Cc);
    const int  b   = (int)(ct / Cc);
    const int  h   = c >> 6;
    const int  d   = c & 63;
    const long bh  = (long)b * Hh + h;
    const float* Pr = P + (bh * Tt + t) * (long)Tt;
    float acc = 0.f;
#pragma unroll
    for (int r = 0; r < 2 * WIN + 1; ++r) {
        const int s = t + r - WIN;
        if (s >= 0 && s < Tt) acc += Pr[s] * embv[r * Dd + d];
    }
    attn[idx] += acc;
}

// ---------------------------------------------------------------------------
// LayerNorm over channel dim of (B,C,T):  h = LN(h + y) * g + b
// One thread per (b,t); loads coalesced along t.
// ---------------------------------------------------------------------------
__global__ void __launch_bounds__(128)
ln_kernel(float* __restrict__ h, const float* __restrict__ y,
          const float* __restrict__ g, const float* __restrict__ be)
{
    const int tid = blockIdx.x * 128 + threadIdx.x;   // B*T
    const int b   = tid / Tt;
    const int t   = tid - b * Tt;
    float* hp = h + (long)b * Cc * Tt + t;
    const float* yp = y + (long)b * Cc * Tt + t;
    float sum = 0.f, sq = 0.f;
    for (int c = 0; c < Cc; ++c) {
        const float v = hp[(long)c * Tt] + yp[(long)c * Tt];
        sum += v; sq += v * v;
    }
    const float mean = sum * (1.f / Cc);
    const float var  = sq * (1.f / Cc) - mean * mean;
    const float rinv = rsqrtf(var + 1e-5f);
    for (int c = 0; c < Cc; ++c) {
        const float v = hp[(long)c * Tt] + yp[(long)c * Tt];
        hp[(long)c * Tt] = (v - mean) * rinv * g[c] + be[c];
    }
}

// ---------------------------------------------------------------------------
// buf = act(buf + bias[c]) * mask[b,t]
// ---------------------------------------------------------------------------
__global__ void bias_act_kernel(float* __restrict__ buf, const float* __restrict__ bias,
                                const float* __restrict__ mask, int Mch, int relu)
{
    const long idx = (long)blockIdx.x * blockDim.x + threadIdx.x;  // B*Mch*T
    const int  t   = (int)(idx % Tt);
    const long rr  = idx / Tt;
    const int  c   = (int)(rr % Mch);
    const int  b   = (int)(rr / Mch);
    float v = buf[idx] + bias[c];
    if (relu) v = fmaxf(v, 0.f);
    buf[idx] = v * mask[b * Tt + t];
}

// ---------------------------------------------------------------------------
// dst = src * mask[b,t]   (over B*C*T)
// ---------------------------------------------------------------------------
__global__ void mask_mul_kernel(float* __restrict__ dst, const float* __restrict__ src,
                                const float* __restrict__ mask)
{
    const long idx = (long)blockIdx.x * blockDim.x + threadIdx.x;
    const int  t   = (int)(idx % Tt);
    const int  b   = (int)(idx / ((long)Cc * Tt));
    dst[idx] = src[idx] * mask[b * Tt + t];
}

// ---------------------------------------------------------------------------

extern "C" void kernel_launch(void* const* d_in, const int* in_sizes, int n_in,
                              void* d_out, int out_size, void* d_ws, size_t ws_size,
                              hipStream_t stream)
{
    (void)in_sizes; (void)n_in; (void)out_size; (void)ws_size;

    const float* x    = (const float*)d_in[0];
    const float* mask = (const float*)d_in[1];
    const float* Wq   = (const float*)d_in[2];
    const float* bq   = (const float*)d_in[3];
    const float* Wk   = (const float*)d_in[4];
    const float* bk   = (const float*)d_in[5];
    const float* Wv   = (const float*)d_in[6];
    const float* bv   = (const float*)d_in[7];
    const float* Wo   = (const float*)d_in[8];
    const float* bo   = (const float*)d_in[9];
    const float* relk = (const float*)d_in[10];
    const float* relv = (const float*)d_in[11];
    const float* n1g  = (const float*)d_in[12];
    const float* n1b  = (const float*)d_in[13];
    const float* n2g  = (const float*)d_in[14];
    const float* n2b  = (const float*)d_in[15];
    const float* fw1  = (const float*)d_in[16];
    const float* fb1  = (const float*)d_in[17];
    const float* fw2  = (const float*)d_in[18];
    const float* fb2  = (const float*)d_in[19];

    float* ws = (float*)d_ws;
    const long SZ = (long)Bb * Cc * Tt;           // 2,097,152 floats
    const long CT = (long)Cc * Tt;
    float* Hb  = ws;                              // hidden state
    float* Qb  = ws + 1 * SZ;                     // q (reused as masked ffn in)
    float* Kb  = ws + 2 * SZ;
    float* Vb  = ws + 3 * SZ;
    float* At  = ws + 4 * SZ;                     // attention context
    float* Yb  = ws + 5 * SZ;                     // branch output
    float* Sb  = ws + 6 * SZ;                     // scores/probs (B*H*T*T)
    float* Rb  = Sb + (long)Bb * Hh * Tt * Tt;    // rel logits band (B*H*T*9)
    float* FFH = Sb;                              // ffn hidden aliases scores

    const dim3 blk(128);
    const dim3 gProj(Tt / 64, Cc / 64, Bb);
    const dim3 gScore(Tt / 64, Tt / 64, Bb * Hh);
    const dim3 gPV(Tt / 64, 1, Bb * Hh);
    const dim3 gF1(Tt / 64, Ff / 64, Bb);
    const dim3 gF2(Tt / 64, Cc / 64, Bb);
    const unsigned gEl  = (unsigned)(SZ / 256);                  // B*C*T / 256
    const unsigned gElF = (unsigned)(((long)Bb * Ff * Tt) / 256);
    const float scale = 1.0f / 8.0f;              // 1/sqrt(64)

    // h = x * mask
    mask_mul_kernel<<<gEl, 256, 0, stream>>>(Hb, x, mask);

    for (int L = 0; L < LAYERS; ++L) {
        const long wofs = (long)L * Cc * Cc;

        // ---- Q/K/V projections: (1024x1024) x (1024x512), batch 4 ----
        gemm_wmma_f16<Cc, 1, false, false, false><<<gProj, blk, 0, stream>>>(
            Wq + wofs, 0, Hb, CT, 0, Qb, CT, bq + (long)L * Cc, 1.f, Cc, Cc);
        gemm_wmma_f16<Cc, 1, false, false, false><<<gProj, blk, 0, stream>>>(
            Wk + wofs, 0, Hb, CT, 0, Kb, CT, bk + (long)L * Cc, 1.f, Cc, Cc);
        gemm_wmma_f16<Cc, 1, false, false, false><<<gProj, blk, 0, stream>>>(
            Wv + wofs, 0, Hb, CT, 0, Vb, CT, bv + (long)L * Cc, 1.f, Cc, Cc);

        // ---- banded rel-k logits ----
        rel_logits_kernel<<<(Bb * Hh * Tt * 9) / 256, 256, 0, stream>>>(
            Qb, relk + (long)L * (2 * WIN + 1) * Dd, Rb, scale);

        // ---- scores: S[t,s] = scale * sum_d q[t,d] k[s,d]  (Q^T K, 64 heads) ----
        gemm_wmma_f16<Tt, 1, true, false, false><<<gScore, blk, 0, stream>>>(
            Qb, (long)Dd * Tt, Kb, (long)Dd * Tt, 0,
            Sb, (long)Tt * Tt, nullptr, scale, Tt, Dd);

        // ---- softmax (+rel band, +mask) ----
        softmax_rel_kernel<<<(Bb * Hh * Tt) / 8, 256, 0, stream>>>(Sb, Rb, mask);

        // ---- context: out[d,t] = sum_s v[d,s] P[t,s]  (V x P^T, 64 heads) ----
        gemm_wmma_f16<Tt, 1, false, true, false><<<gPV, blk, 0, stream>>>(
            Vb, (long)Dd * Tt, Sb, (long)Tt * Tt, 0,
            At, (long)Dd * Tt, nullptr, 1.f, Dd, Tt);

        // ---- + rel-v band ----
        add_relv_kernel<<<gEl, 256, 0, stream>>>(At, Sb, relv + (long)L * (2 * WIN + 1) * Dd);

        // ---- output projection ----
        gemm_wmma_f16<Cc, 1, false, false, false><<<gProj, blk, 0, stream>>>(
            Wo + wofs, 0, At, CT, 0, Yb, CT, bo + (long)L * Cc, 1.f, Cc, Cc);

        // ---- LN1: h = LN(h + y) ----
        ln_kernel<<<(Bb * Tt) / 128, 128, 0, stream>>>(Hb, Yb, n1g + (long)L * Cc, n1b + (long)L * Cc);

        // ---- FFN input: masked copy (reuse Qb) ----
        mask_mul_kernel<<<gEl, 256, 0, stream>>>(Qb, Hb, mask);

        // ---- conv1 (1024 -> 4096, KS=3) as 3 shifted accumulating GEMMs ----
        for (int kk = 0; kk < KSc; ++kk) {
            const float* wA = fw1 + (long)L * Ff * Cc * KSc + kk;
            if (kk == 0)
                gemm_wmma_f16<Cc * KSc, KSc, false, false, false><<<gF1, blk, 0, stream>>>(
                    wA, 0, Qb, CT, kk - 1, FFH, (long)Ff * Tt, nullptr, 1.f, Ff, Cc);
            else
                gemm_wmma_f16<Cc * KSc, KSc, false, false, true><<<gF1, blk, 0, stream>>>(
                    wA, 0, Qb, CT, kk - 1, FFH, (long)Ff * Tt, nullptr, 1.f, Ff, Cc);
        }
        bias_act_kernel<<<gElF, 256, 0, stream>>>(FFH, fb1 + (long)L * Ff, mask, Ff, 1);

        // ---- conv2 (4096 -> 1024, KS=3) ----
        for (int kk = 0; kk < KSc; ++kk) {
            const float* wA = fw2 + (long)L * Cc * Ff * KSc + kk;
            if (kk == 0)
                gemm_wmma_f16<Ff * KSc, KSc, false, false, false><<<gF2, blk, 0, stream>>>(
                    wA, 0, FFH, (long)Ff * Tt, kk - 1, Yb, CT, nullptr, 1.f, Cc, Ff);
            else
                gemm_wmma_f16<Ff * KSc, KSc, false, false, true><<<gF2, blk, 0, stream>>>(
                    wA, 0, FFH, (long)Ff * Tt, kk - 1, Yb, CT, nullptr, 1.f, Cc, Ff);
        }
        bias_act_kernel<<<gEl, 256, 0, stream>>>(Yb, fb2 + (long)L * Cc, mask, Cc, 0);

        // ---- LN2: h = LN(h + y) ----
        ln_kernel<<<(Bb * Tt) / 128, 128, 0, stream>>>(Hb, Yb, n2g + (long)L * Cc, n2b + (long)L * Cc);
    }

    // out = h * mask
    mask_mul_kernel<<<gEl, 256, 0, stream>>>((float*)d_out, Hb, mask);
}
